// MultiHeadAttention_40046275068333
// MI455X (gfx1250) — matrix operand order repaired and checked by reading, not compiled, not run on hardware
//
#include <hip/hip_runtime.h>
#include <hip/hip_bf16.h>

typedef __attribute__((ext_vector_type(16))) _Float16 v16h;
typedef __attribute__((ext_vector_type(8)))  _Float16 v8h;
typedef __attribute__((ext_vector_type(8)))  float    v8f;

union F8 { v8f v; float f[8]; };

#define SEQ   2048
#define EMB   1024
#define HEADS 16
#define HDIM  64
#define BATCH 2
#define MTOT  (BATCH*SEQ)   // 4096

__device__ __forceinline__ v16h cat8(v8h lo, v8h hi) {
  return __builtin_shufflevector(lo, hi, 0,1,2,3,4,5,6,7,8,9,10,11,12,13,14,15);
}

// ---------------------------------------------------------------------------
// One-time f32 -> f16 conversion (vectorized: 8 elems/thread).
// ---------------------------------------------------------------------------
__global__ __launch_bounds__(256) void cvt_f32_to_f16(
    const float* __restrict__ in, _Float16* __restrict__ out, int n)
{
  const int i = (blockIdx.x * 256 + threadIdx.x) * 8;
  if (i >= n) return;
  float4 f0 = ((const float4*)(in + i))[0];
  float4 f1 = ((const float4*)(in + i))[1];
  v8h o;
  o[0] = (_Float16)f0.x; o[1] = (_Float16)f0.y;
  o[2] = (_Float16)f0.z; o[3] = (_Float16)f0.w;
  o[4] = (_Float16)f1.x; o[5] = (_Float16)f1.y;
  o[6] = (_Float16)f1.z; o[7] = (_Float16)f1.w;
  *(v8h*)(out + i) = o;
}

// ---------------------------------------------------------------------------
// Y[m,n] = sum_k X[m,k] * W[n,k]  (X @ W^T), f16 in, f32 WMMA accum.
// K = N = EMB fixed at compile time; MODE is a template param so each
// instantiation keeps exactly one store path (kills runtime address math).
// Wave computes a 16(M) x 64(N) strip; block = 8 waves = 8 M-tiles.
// MODE: 0 -> Y[M,EMB] row-major
//       1 -> [B,H,S,D] head layout
//       2 -> [B,H,D,S] transposed head layout
// ---------------------------------------------------------------------------
template <typename OutT, int MODE>
__global__ __launch_bounds__(256) void wmma_gemm_xwT(
    const _Float16* __restrict__ X, const _Float16* __restrict__ W,
    OutT* __restrict__ Y)
{
  const int lane = threadIdx.x & 31;
  const int wid  = threadIdx.x >> 5;
  const int col  = lane & 15;
  const int half = lane >> 4;

  const int mbase = (blockIdx.y * 8 + wid) * 16;
  const int nbase = blockIdx.x * 64;

  const _Float16* xrow = X + (size_t)(mbase + col) * EMB;

  v8f acc[4];
  #pragma unroll
  for (int t = 0; t < 4; ++t) acc[t] = (v8f){0,0,0,0,0,0,0,0};

  for (int kb = 0; kb < EMB; kb += 32) {
    v8h a0 = *(const v8h*)(xrow + kb +     16 * half);
    v8h a1 = *(const v8h*)(xrow + kb + 8 + 16 * half);
    v8h blo[4], bhi[4];
    #pragma unroll
    for (int t = 0; t < 4; ++t) {
      const _Float16* wp = W + (size_t)(nbase + t * 16 + col) * EMB + kb + 16 * half;
      blo[t] = *(const v8h*)wp;
      bhi[t] = *(const v8h*)(wp + 8);
    }
    const v16h a = cat8(a0, a1);
    #pragma unroll
    for (int t = 0; t < 4; ++t)
      acc[t] = __builtin_amdgcn_wmma_f32_16x16x32_f16(
          false, a, false, cat8(blo[t], bhi[t]), (short)0, acc[t], false, false);
  }

  #pragma unroll
  for (int t = 0; t < 4; ++t) {
    F8 c; c.v = acc[t];
    const int gn = nbase + t * 16 + col;
    #pragma unroll
    for (int r = 0; r < 8; ++r) {
      const int gm = mbase + r + 8 * half;
      size_t off;
      if (MODE == 0) {
        off = (size_t)gm * EMB + gn;
      } else {
        const int b = gm >> 11, s = gm & (SEQ - 1);
        const int h = gn >> 6,  d = gn & (HDIM - 1);
        if (MODE == 1) off = ((size_t)(b * HEADS + h) * SEQ + s) * HDIM + d;
        else           off = ((size_t)(b * HEADS + h) * HDIM + d) * SEQ + s;
      }
      Y[off] = (OutT)c.f[r];
    }
  }
}

// ---------------------------------------------------------------------------
// Flash attention, all-f16 operands. Q,K in [B,H,S,D] f16; Vt in [B,H,D,S]
// f16; AO out token-major [B,S,E] f16. Softmax state in f32.
// Row sums computed by an extra WMMA against an all-ones B matrix
// (C[m,n] = sum_k P[m,k]); only the row-max needs a shuffle reduction.
// ---------------------------------------------------------------------------
__global__ __launch_bounds__(256) void wmma_flash_attn(
    const _Float16* __restrict__ Q, const _Float16* __restrict__ Kmat,
    const _Float16* __restrict__ Vt, const int* __restrict__ mask,
    _Float16* __restrict__ AO)
{
  __shared__ _Float16 plds[8 * 16 * 32];   // 1 KB per wave

  const int lane = threadIdx.x & 31;
  const int wid  = threadIdx.x >> 5;
  const int col  = lane & 15;
  const int half = lane >> 4;
  _Float16* pbuf = plds + wid * (16 * 32);

  const int gw = blockIdx.x * 8 + wid;        // 4096 waves total
  const int bh = gw >> 7;
  const int qb = gw & 127;
  const int b  = bh >> 4;
  const int h  = bh & 15;

  // Constant all-ones B fragment for row-sum WMMA
  v16h onesf;
  #pragma unroll
  for (int i = 0; i < 16; ++i) onesf[i] = (_Float16)1.0f;

  // Q A-fragments for the two 32-wide d-chunks
  const _Float16* qrow = Q + ((size_t)bh * SEQ + qb * 16 + col) * HDIM;
  const v16h aq0 = cat8(*(const v8h*)(qrow +      16 * half),
                        *(const v8h*)(qrow +  8 + 16 * half));
  const v16h aq1 = cat8(*(const v8h*)(qrow + 32 + 16 * half),
                        *(const v8h*)(qrow + 40 + 16 * half));

  float rmax[8], rsum[8], corr[8];
  F8 acc[4];
  #pragma unroll
  for (int r = 0; r < 8; ++r) {
    rmax[r] = -INFINITY; rsum[r] = 0.0f;
    acc[0].f[r] = 0.0f; acc[1].f[r] = 0.0f; acc[2].f[r] = 0.0f; acc[3].f[r] = 0.0f;
  }

  const _Float16* kbase = Kmat + (size_t)bh * SEQ * HDIM;
  const _Float16* vbase = Vt   + (size_t)bh * HDIM * SEQ;
  const int* mrow = mask + b * SEQ;
  const int iqbase = qb * 16 + 8 * half;
  const float kscale = 0.125f;              // 1/sqrt(64)
  const float bscale = 0.5f / (float)SEQ;   // ALPHA / S

  for (int j = 0; j < SEQ / 32; ++j) {
    const int k0 = j * 32;

    // ---- batch all K-fragment loads, then QK^T WMMAs ----
    v8h kf[8];
    #pragma unroll
    for (int t = 0; t < 2; ++t) {
      const _Float16* krow = kbase + (size_t)(k0 + t * 16 + col) * HDIM;
      kf[t*4+0] = *(const v8h*)(krow +      16 * half);
      kf[t*4+1] = *(const v8h*)(krow +  8 + 16 * half);
      kf[t*4+2] = *(const v8h*)(krow + 32 + 16 * half);
      kf[t*4+3] = *(const v8h*)(krow + 40 + 16 * half);
    }
    F8 sc[2];
    #pragma unroll
    for (int t = 0; t < 2; ++t) {
      v8f ct = (v8f){0,0,0,0,0,0,0,0};
      ct = __builtin_amdgcn_wmma_f32_16x16x32_f16(
          false, aq0, false, cat8(kf[t*4+0], kf[t*4+1]), (short)0, ct, false, false);
      ct = __builtin_amdgcn_wmma_f32_16x16x32_f16(
          false, aq1, false, cat8(kf[t*4+2], kf[t*4+3]), (short)0, ct, false, false);
      sc[t].v = ct;
    }

    const int kk0 = k0 + col, kk1 = kk0 + 16;
    const int m0 = mrow[kk0], m1 = mrow[kk1];

    // ---- scale + mirror bias + mask, online max, exp, acc rescale ----
    #pragma unroll
    for (int r = 0; r < 8; ++r) {
      const float target = (float)(SEQ - 1 - (iqbase + r));
      float s0 = sc[0].f[r] * kscale - bscale * fabsf((float)kk0 - target);
      float s1 = sc[1].f[r] * kscale - bscale * fabsf((float)kk1 - target);
      if (m0) s0 = -1e30f;
      if (m1) s1 = -1e30f;
      float tm = fmaxf(s0, s1);
      #pragma unroll
      for (int o = 1; o < 16; o <<= 1) tm = fmaxf(tm, __shfl_xor(tm, o, 16));
      const float mnew = fmaxf(rmax[r], tm);
      corr[r] = __expf(rmax[r] - mnew);
      rmax[r] = mnew;
      const float p0 = __expf(s0 - mnew);
      const float p1 = __expf(s1 - mnew);
      acc[0].f[r] *= corr[r]; acc[1].f[r] *= corr[r];
      acc[2].f[r] *= corr[r]; acc[3].f[r] *= corr[r];
      // relayout P (C-layout) -> LDS tile for A-layout reload
      const int m = r + 8 * half;
      pbuf[m * 32 + col]      = (_Float16)p0;
      pbuf[m * 32 + 16 + col] = (_Float16)p1;
    }

    // same-wave LDS ops are in-order (DScnt); no barrier needed
    const v16h pfrag = cat8(*(const v8h*)(pbuf + col * 32 +     16 * half),
                            *(const v8h*)(pbuf + col * 32 + 8 + 16 * half));

    // ---- row sums of P via WMMA against all-ones B ----
    F8 sums;
    sums.v = __builtin_amdgcn_wmma_f32_16x16x32_f16(
        false, pfrag, false, onesf, (short)0,
        (v8f){0,0,0,0,0,0,0,0}, false, false);
    #pragma unroll
    for (int r = 0; r < 8; ++r)
      rsum[r] = rsum[r] * corr[r] + sums.f[r];

    // ---- batch all V-fragment loads, then P@V WMMAs ----
    v8h vf[8];
    #pragma unroll
    for (int dt = 0; dt < 4; ++dt) {
      const _Float16* vp = vbase + (size_t)(dt * 16 + col) * SEQ + k0 + 16 * half;
      vf[dt*2+0] = *(const v8h*)vp;
      vf[dt*2+1] = *(const v8h*)(vp + 8);
    }
    #pragma unroll
    for (int dt = 0; dt < 4; ++dt)
      acc[dt].v = __builtin_amdgcn_wmma_f32_16x16x32_f16(
          false, pfrag, false, cat8(vf[dt*2+0], vf[dt*2+1]),
          (short)0, acc[dt].v, false, false);
  }

  // ---- normalize, store f16 token-major [B,S,E] for the O projection ----
  #pragma unroll
  for (int r = 0; r < 8; ++r) {
    const float inv = 1.0f / rsum[r];
    const int sq = qb * 16 + r + 8 * half;
    _Float16* orow = AO + ((size_t)b * SEQ + sq) * EMB + h * HDIM;
    #pragma unroll
    for (int dt = 0; dt < 4; ++dt)
      orow[dt * 16 + col] = (_Float16)(acc[dt].f[r] * inv);
  }
}

// ---------------------------------------------------------------------------
extern "C" void kernel_launch(void* const* d_in, const int* in_sizes, int n_in,
                              void* d_out, int out_size, void* d_ws, size_t ws_size,
                              hipStream_t stream) {
  (void)in_sizes; (void)n_in; (void)out_size; (void)ws_size;
  const float* x    = (const float*)d_in[0];
  const float* Wq   = (const float*)d_in[1];
  const float* Wk   = (const float*)d_in[2];
  const float* Wv   = (const float*)d_in[3];
  const float* Wo   = (const float*)d_in[4];
  const int*   mask = (const int*)d_in[5];
  float* out = (float*)d_out;

  const size_t nx = (size_t)MTOT * EMB;   // 4M elements
  const size_t nw = (size_t)EMB * EMB;    // 1M elements
  _Float16* xh  = (_Float16*)d_ws;
  _Float16* Wqh = xh  + nx;
  _Float16* Wkh = Wqh + nw;
  _Float16* Wvh = Wkh + nw;
  _Float16* Woh = Wvh + nw;
  _Float16* Qh  = Woh + nw;
  _Float16* Kh  = Qh  + nx;
  _Float16* Vth = Kh  + nx;
  _Float16* AOh = Vth + nx;               // total ~48 MB of ws

  dim3 blk(256);
  cvt_f32_to_f16<<<dim3(nx / (8 * 256)), blk, 0, stream>>>(x,  xh,  (int)nx);
  cvt_f32_to_f16<<<dim3(nw / (8 * 256)), blk, 0, stream>>>(Wq, Wqh, (int)nw);
  cvt_f32_to_f16<<<dim3(nw / (8 * 256)), blk, 0, stream>>>(Wk, Wkh, (int)nw);
  cvt_f32_to_f16<<<dim3(nw / (8 * 256)), blk, 0, stream>>>(Wv, Wvh, (int)nw);
  cvt_f32_to_f16<<<dim3(nw / (8 * 256)), blk, 0, stream>>>(Wo, Woh, (int)nw);

  dim3 ggrid(EMB / 64, MTOT / (8 * 16));   // (16, 32)
  wmma_gemm_xwT<_Float16, 1><<<ggrid, blk, 0, stream>>>(xh, Wqh, Qh);
  wmma_gemm_xwT<_Float16, 1><<<ggrid, blk, 0, stream>>>(xh, Wkh, Kh);
  wmma_gemm_xwT<_Float16, 2><<<ggrid, blk, 0, stream>>>(xh, Wvh, Vth);

  wmma_flash_attn<<<dim3((BATCH * HEADS * (SEQ / 16)) / 8), blk, 0, stream>>>(
      Qh, Kh, Vth, mask, AOh);

  wmma_gemm_xwT<float, 0><<<ggrid, blk, 0, stream>>>(AOh, Woh, out);
}
